// Predicate2EntityCrossAttentionEncoder_88064009437724
// MI455X (gfx1250) — matrix operand order repaired and checked by reading, not compiled, not run on hardware
//
#include <hip/hip_runtime.h>

#define DM 512
#define DI 2048
#define NB 32

typedef __attribute__((ext_vector_type(16))) _Float16 v16h;
typedef __attribute__((ext_vector_type(8)))  _Float16 v8h;
typedef __attribute__((ext_vector_type(8)))  float    v8f;

static __device__ __forceinline__ int imin(int a, int b) { return a < b ? a : b; }

static __device__ __forceinline__ v8f wmma_f16(v16h a, v16h b, v8f c) {
  return __builtin_amdgcn_wmma_f32_16x16x32_f16(false, a, false, b, (short)0, c, false, false);
}

// ---------------- f32 -> f16 convert ----------------
__global__ __launch_bounds__(256) void k_cvt(const float* __restrict__ s,
                                             _Float16* __restrict__ d, int n) {
  int i = blockIdx.x * 256 + threadIdx.x;
  if (i < n) d[i] = (_Float16)s[i];
}

// ---------------- WMMA GEMM: C[M,N] = A[M,K] @ W[N,K]^T + bias ----------------
// Block tile: 32 rows x 512 cols; wave tile 32x64 (2 M-subtiles x 4 N-tiles).
// Double-buffered LDS A staging -> one barrier per 32-wide K chunk.
// mode 0: f32 out, 1: f16 out, 2: f16 relu out, 3: f16 out transposed per-head (V -> Vt)
__global__ __launch_bounds__(256) void k_gemm(const _Float16* __restrict__ A,
                                              const _Float16* __restrict__ W,
                                              const float* __restrict__ bias,
                                              void* __restrict__ out,
                                              int M, int N, int K, int mode,
                                              int Lv, int LvP) {
  __shared__ _Float16 atile[2][32 * 40];  // 32 rows x 32 cols, padded stride 40 (80B)
  const int tid  = threadIdx.x;
  const int lane = tid & 31, wave = tid >> 5;
  const int lm   = lane & 15, hi = lane >> 4;
  const int m0   = blockIdx.y * 32;
  const int n_wave = blockIdx.x * 512 + wave * 64;

  const v8f z = {0.f,0.f,0.f,0.f,0.f,0.f,0.f,0.f};
  v8f acc[2][4] = {{z, z, z, z}, {z, z, z, z}};

  const int ar = tid >> 3;          // staging row 0..31
  const int ac = (tid & 7) * 4;     // staging col (4 f16 per thread, 8B copy)
  const size_t arow = (size_t)(m0 + ar) * K;

  // stage first chunk
  *(unsigned long long*)&atile[0][ar * 40 + ac] =
      *(const unsigned long long*)(A + arow + ac);
  __syncthreads();

  int cur = 0;
  for (int k0 = 0; k0 < K; k0 += 32) {
    // stage next chunk into the other buffer (overlaps with this chunk's math)
    if (k0 + 32 < K) {
      *(unsigned long long*)&atile[cur ^ 1][ar * 40 + ac] =
          *(const unsigned long long*)(A + arow + k0 + 32 + ac);
      __builtin_prefetch(A + arow + k0 + 64 + ac, 0, 3);
    }
    // batch all 4 B tiles (16 contiguous K per lane, 32B aligned)
    v16h b[4];
#pragma unroll
    for (int t = 0; t < 4; ++t)
      b[t] = *(const v16h*)(W + (size_t)(n_wave + t * 16 + lm) * K + k0 + hi * 16);
    // two A subtiles from LDS
    v16h a0, a1;
    {
      v8h* p0 = (v8h*)&a0;
      p0[0] = *(const v8h*)&atile[cur][lm * 40 + hi * 16];
      p0[1] = *(const v8h*)&atile[cur][lm * 40 + hi * 16 + 8];
      v8h* p1 = (v8h*)&a1;
      p1[0] = *(const v8h*)&atile[cur][(16 + lm) * 40 + hi * 16];
      p1[1] = *(const v8h*)&atile[cur][(16 + lm) * 40 + hi * 16 + 8];
    }
#pragma unroll
    for (int t = 0; t < 4; ++t) acc[0][t] = wmma_f16(a0, b[t], acc[0][t]);
#pragma unroll
    for (int t = 0; t < 4; ++t) acc[1][t] = wmma_f16(a1, b[t], acc[1][t]);
    __syncthreads();
    cur ^= 1;
  }

#pragma unroll
  for (int mt = 0; mt < 2; ++mt) {
#pragma unroll
    for (int t = 0; t < 4; ++t) {
      const int n = n_wave + t * 16 + lm;
      const float bv = bias[n];
#pragma unroll
      for (int r = 0; r < 8; ++r) {
        const int m = m0 + mt * 16 + r + 8 * hi;
        float v = acc[mt][t][r] + bv;
        if (mode == 0) {
          ((float*)out)[(size_t)m * N + n] = v;
        } else if (mode == 1) {
          ((_Float16*)out)[(size_t)m * N + n] = (_Float16)v;
        } else if (mode == 2) {
          ((_Float16*)out)[(size_t)m * N + n] = (_Float16)fmaxf(v, 0.f);
        } else {
          const int bidx = m / Lv, kv = m - bidx * Lv;
          const int h = n >> 6, dd = n & 63;
          ((_Float16*)out)[(((size_t)bidx * 8 + h) * 64 + dd) * LvP + kv] = (_Float16)v;
        }
      }
    }
  }
}

// ---------------- flash attention: one wave per (b, h, 16-query tile) ----------------
__global__ __launch_bounds__(256) void k_attn(const _Float16* __restrict__ Q,
                                              const _Float16* __restrict__ Kr,
                                              const _Float16* __restrict__ Vt,
                                              const unsigned char* __restrict__ msk,
                                              _Float16* __restrict__ ctx,
                                              int Lq, int Lv, int LvP, int qtiles) {
  __shared__ _Float16 plds[8 * 16 * 40];  // per-wave 16x32 P tile, padded stride 40
  const int lane = threadIdx.x & 31, wave = threadIdx.x >> 5;
  const int lm = lane & 15, hi = lane >> 4;
  const int u = blockIdx.x * 8 + wave;
  const int per_b = 8 * qtiles;
  const int b  = u / per_b;
  const int rm = u - b * per_b;
  const int h  = rm / qtiles;
  const int qt = rm - h * qtiles;
  const int q0 = qt * 16;
  _Float16* pl = &plds[wave * 16 * 40];

  // Q A-tiles (K-dim = d_k = 64 -> two 16x32 tiles), loaded once
  const int qrow = imin(q0 + lm, Lq - 1);
  const _Float16* qp = Q + ((size_t)b * Lq + qrow) * DM + h * 64;
  v16h qa0, qa1;
  {
    v8h* t0 = (v8h*)&qa0; t0[0] = *(const v8h*)(qp + hi * 16);      t0[1] = *(const v8h*)(qp + hi * 16 + 8);
    v8h* t1 = (v8h*)&qa1; t1[0] = *(const v8h*)(qp + 32 + hi * 16); t1[1] = *(const v8h*)(qp + 40 + hi * 16);
  }

  const v8f z = {0.f,0.f,0.f,0.f,0.f,0.f,0.f,0.f};
  v8f acc[4] = {z, z, z, z};
  float rmax[8], rsum[8];
#pragma unroll
  for (int r = 0; r < 8; ++r) { rmax[r] = -1e30f; rsum[r] = 0.f; }

  const _Float16* vbase = Vt + ((size_t)b * 8 + h) * 64 * LvP;

  for (int kv0 = 0; kv0 < Lv; kv0 += 32) {
    // ---- batch-load K tiles, then 4 score WMMAs ----
    v16h kb[2][2];
#pragma unroll
    for (int t = 0; t < 2; ++t) {
      const int kvi = imin(kv0 + t * 16 + lm, Lv - 1);
      const _Float16* kp = Kr + ((size_t)b * Lv + kvi) * DM + h * 64 + hi * 16;
      kb[t][0] = *(const v16h*)kp;         // d = hi*16..+15
      kb[t][1] = *(const v16h*)(kp + 32);  // d = 32+hi*16..+15
    }
    v8f s[2] = {z, z};
#pragma unroll
    for (int t = 0; t < 2; ++t) {
      s[t] = wmma_f16(qa0, kb[t][0], s[t]);
      s[t] = wmma_f16(qa1, kb[t][1], s[t]);
    }
    // ---- issue V tiles early (independent of softmax) ----
    v16h vb[4];
#pragma unroll
    for (int t4 = 0; t4 < 4; ++t4)
      vb[t4] = *(const v16h*)(vbase + (size_t)(t4 * 16 + lm) * LvP + kv0 + hi * 16);
    // ---- scale + mask ----
#pragma unroll
    for (int r = 0; r < 8; ++r) {
      const int q  = q0 + r + 8 * hi;
      const int mq = imin(q, Lq - 1);
#pragma unroll
      for (int t = 0; t < 2; ++t) {
        const int kv = kv0 + t * 16 + lm;
        float sv = s[t][r] * 0.125f;  // 1/sqrt(64)
        const bool bad = (kv >= Lv) || (msk[((size_t)b * Lq + mq) * Lv + kv] != 0);
        s[t][r] = bad ? -3.0e38f : sv;
      }
    }
    // ---- online softmax (row stats replicated across each 16-lane group) ----
#pragma unroll
    for (int r = 0; r < 8; ++r) {
      float cm = fmaxf(s[0][r], s[1][r]);
#pragma unroll
      for (int o = 1; o < 16; o <<= 1) cm = fmaxf(cm, __shfl_xor(cm, o, 32));
      const float nm = fmaxf(rmax[r], cm);
      const float corr = __expf(rmax[r] - nm);
      rmax[r] = nm;
      float ps = 0.f;
#pragma unroll
      for (int t = 0; t < 2; ++t) {
        const float pv = __expf(s[t][r] - nm);
        s[t][r] = pv;
        ps += pv;
      }
#pragma unroll
      for (int o = 1; o < 16; o <<= 1) ps += __shfl_xor(ps, o, 32);
      rsum[r] = rsum[r] * corr + ps;
#pragma unroll
      for (int t4 = 0; t4 < 4; ++t4) acc[t4][r] *= corr;
    }
    // ---- re-layout P (D-layout -> A-layout) through LDS ----
#pragma unroll
    for (int t = 0; t < 2; ++t)
#pragma unroll
      for (int r = 0; r < 8; ++r)
        pl[(r + 8 * hi) * 40 + t * 16 + lm] = (_Float16)s[t][r];
    __syncthreads();
    v16h pa;
    {
      v8h* ph = (v8h*)&pa;
      ph[0] = *(const v8h*)&pl[lm * 40 + hi * 16];
      ph[1] = *(const v8h*)&pl[lm * 40 + hi * 16 + 8];
    }
    // ---- ctx += P @ V ----
#pragma unroll
    for (int t4 = 0; t4 < 4; ++t4) acc[t4] = wmma_f16(pa, vb[t4], acc[t4]);
  }

  // ---- normalize + store f16 ctx ----
#pragma unroll
  for (int t4 = 0; t4 < 4; ++t4) {
#pragma unroll
    for (int r = 0; r < 8; ++r) {
      const int q = q0 + r + 8 * hi;
      if (q < Lq) {
        const float v = acc[t4][r] / rsum[r];
        ctx[((size_t)b * Lq + q) * DM + h * 64 + t4 * 16 + lm] = (_Float16)v;
      }
    }
  }
}

// ---------------- fused residual-add + LayerNorm + pad-mask ----------------
__global__ __launch_bounds__(256) void k_addln(const float* __restrict__ x,
                                               const float* __restrict__ res,
                                               const float* __restrict__ g,
                                               const float* __restrict__ be,
                                               const unsigned char* __restrict__ npm,
                                               float* __restrict__ o32,
                                               _Float16* __restrict__ o16) {
  __shared__ float red[16];
  const int row = blockIdx.x;
  const int tid = threadIdx.x;
  const size_t base = (size_t)row * DM;
  const float v0 = x[base + tid] + res[base + tid];
  const float v1 = x[base + 256 + tid] + res[base + 256 + tid];
  float s = v0 + v1, q = v0 * v0 + v1 * v1;
#pragma unroll
  for (int o = 1; o < 32; o <<= 1) { s += __shfl_xor(s, o, 32); q += __shfl_xor(q, o, 32); }
  const int wave = tid >> 5, lane = tid & 31;
  if (lane == 0) { red[wave] = s; red[8 + wave] = q; }
  __syncthreads();
  float ts = 0.f, tq = 0.f;
#pragma unroll
  for (int i = 0; i < 8; ++i) { ts += red[i]; tq += red[8 + i]; }
  const float mean = ts * (1.f / 512.f);
  const float var  = tq * (1.f / 512.f) - mean * mean;
  const float rstd = rsqrtf(var + 1e-5f);
  const float np = npm[row] ? 1.f : 0.f;
  const float y0 = ((v0 - mean) * rstd * g[tid] + be[tid]) * np;
  const float y1 = ((v1 - mean) * rstd * g[tid + 256] + be[tid + 256]) * np;
  o32[base + tid] = y0;
  o32[base + 256 + tid] = y1;
  if (o16) { o16[base + tid] = (_Float16)y0; o16[base + 256 + tid] = (_Float16)y1; }
}

// ---------------- gather valid rows ----------------
__global__ __launch_bounds__(256) void k_gather(const float* __restrict__ f,
                                                const int* __restrict__ idx,
                                                float* __restrict__ out) {
  const int row = blockIdx.x, tid = threadIdx.x;
  const size_t src = (size_t)idx[row] * DM;
  const size_t dst = (size_t)row * DM;
  out[dst + tid] = f[src + tid];
  out[dst + 256 + tid] = f[src + 256 + tid];
}

extern "C" void kernel_launch(void* const* d_in, const int* in_sizes, int n_in,
                              void* d_out, int out_size, void* d_ws, size_t ws_size,
                              hipStream_t stream) {
  (void)n_in; (void)ws_size;
  const float* ent  = (const float*)d_in[0];
  const float* pred = (const float*)d_in[1];
  const float* w_qs = (const float*)d_in[2];
  const float* b_qs = (const float*)d_in[3];
  const float* w_ks = (const float*)d_in[4];
  const float* b_ks = (const float*)d_in[5];
  const float* w_vs = (const float*)d_in[6];
  const float* b_vs = (const float*)d_in[7];
  const float* w_fc = (const float*)d_in[8];
  const float* b_fc = (const float*)d_in[9];
  const float* ln1g = (const float*)d_in[10];
  const float* ln1b = (const float*)d_in[11];
  const float* w1   = (const float*)d_in[12];
  const float* b1   = (const float*)d_in[13];
  const float* w2   = (const float*)d_in[14];
  const float* b2   = (const float*)d_in[15];
  const float* ln2g = (const float*)d_in[16];
  const float* ln2b = (const float*)d_in[17];
  const unsigned char* amask = (const unsigned char*)d_in[18];
  const unsigned char* npm   = (const unsigned char*)d_in[19];
  const int* gidx = (const int*)d_in[20];

  const int Lq = in_sizes[0] / (NB * DM);
  const int Lv = in_sizes[1] / (NB * DM);
  const int LvP = (Lv + 15) & ~15;
  const int Mq = NB * Lq, Mv = NB * Lv;
  const int qtiles = (Lq + 15) / 16;

  char* wsp = (char*)d_ws;
  size_t off = 0;
  auto alloc = [&](size_t bytes) -> char* {
    char* p = wsp + off;
    off = (off + bytes + 255) & ~(size_t)255;
    return p;
  };
  _Float16* ent16  = (_Float16*)alloc((size_t)Mq * DM * 2);
  _Float16* pred16 = (_Float16*)alloc((size_t)Mv * DM * 2);
  _Float16* wq16   = (_Float16*)alloc((size_t)DM * DM * 2);
  _Float16* wk16   = (_Float16*)alloc((size_t)DM * DM * 2);
  _Float16* wv16   = (_Float16*)alloc((size_t)DM * DM * 2);
  _Float16* wfc16  = (_Float16*)alloc((size_t)DM * DM * 2);
  _Float16* w1_16  = (_Float16*)alloc((size_t)DI * DM * 2);
  _Float16* w2_16  = (_Float16*)alloc((size_t)DM * DI * 2);
  _Float16* q16    = (_Float16*)alloc((size_t)Mq * DM * 2);
  _Float16* k16    = (_Float16*)alloc((size_t)Mv * DM * 2);
  _Float16* vt16   = (_Float16*)alloc((size_t)NB * 8 * 64 * LvP * 2);
  _Float16* ctx16  = (_Float16*)alloc((size_t)Mq * DM * 2);
  float*    fc32   = (float*)alloc((size_t)Mq * DM * 4);  // later reused in-place as LN1 out
  _Float16* out116 = (_Float16*)alloc((size_t)Mq * DM * 2);
  _Float16* h16    = (_Float16*)alloc((size_t)Mq * DI * 2);
  float*    f2     = (float*)alloc((size_t)Mq * DM * 4);  // LN2 in-place

  auto cvt = [&](const float* s, _Float16* d, size_t n) {
    k_cvt<<<dim3((unsigned)((n + 255) / 256)), dim3(256), 0, stream>>>(s, d, (int)n);
  };
  cvt(ent, ent16, (size_t)Mq * DM);
  cvt(pred, pred16, (size_t)Mv * DM);
  cvt(w_qs, wq16, (size_t)DM * DM);
  cvt(w_ks, wk16, (size_t)DM * DM);
  cvt(w_vs, wv16, (size_t)DM * DM);
  cvt(w_fc, wfc16, (size_t)DM * DM);
  cvt(w1, w1_16, (size_t)DI * DM);
  cvt(w2, w2_16, (size_t)DM * DI);

  const dim3 blk(256);
  // Q / K / V projections (V stored transposed per head for the PV WMMA B-operand)
  k_gemm<<<dim3(1, Mq / 32), blk, 0, stream>>>(ent16, wq16, b_qs, q16, Mq, DM, DM, 1, 1, 1);
  k_gemm<<<dim3(1, Mv / 32), blk, 0, stream>>>(pred16, wk16, b_ks, k16, Mv, DM, DM, 1, 1, 1);
  k_gemm<<<dim3(1, Mv / 32), blk, 0, stream>>>(pred16, wv16, b_vs, vt16, Mv, DM, DM, 3, Lv, LvP);
  // flash cross-attention
  k_attn<<<dim3(NB * qtiles), blk, 0, stream>>>(q16, k16, vt16, amask, ctx16, Lq, Lv, LvP, qtiles);
  // output projection
  k_gemm<<<dim3(1, Mq / 32), blk, 0, stream>>>(ctx16, wfc16, b_fc, fc32, Mq, DM, DM, 0, 1, 1);
  // residual + LN1 + pad mask (in-place on fc32; f16 copy for FFN)
  k_addln<<<dim3(Mq), blk, 0, stream>>>(fc32, ent, ln1g, ln1b, npm, fc32, out116);
  // FFN
  k_gemm<<<dim3(DI / 512, Mq / 32), blk, 0, stream>>>(out116, w1_16, b1, h16, Mq, DI, DM, 2, 1, 1);
  k_gemm<<<dim3(1, Mq / 32), blk, 0, stream>>>(h16, w2_16, b2, f2, Mq, DM, DI, 0, 1, 1);
  // residual + LN2 + pad mask (in-place)
  k_addln<<<dim3(Mq), blk, 0, stream>>>(f2, fc32, ln2g, ln2b, npm, f2, nullptr);
  // unpad gather
  const int nrows = out_size / DM;
  k_gather<<<dim3(nrows), blk, 0, stream>>>(f2, gidx, (float*)d_out);
}
